// ReBasedLinearAttention_41489384079872
// MI455X (gfx1250) — compile-verified, matrix operand order checked by reading, not run on hardware
//
#include <hip/hip_runtime.h>
#include <hip/hip_bf16.h>

typedef _Float16 f16;
typedef __attribute__((ext_vector_type(16))) _Float16 v16h;
typedef __attribute__((ext_vector_type(8)))  _Float16 h8;
typedef __attribute__((ext_vector_type(4)))  _Float16 h4;
typedef __attribute__((ext_vector_type(8)))  float    v8f;
typedef __attribute__((ext_vector_type(4)))  float    f4;

#define T_SEQ 2048
#define HID   2048
#define NH    16
#define HD    128
#define LN_EPS 1e-5f
#define LDP   40   // padded f16 per 32-k LDS row (80 B -> conflict-free frags)

// ---- CDNA5 async global->LDS copy (ASYNCcnt-tracked, ISA 15.18 op 98) ------
__device__ __forceinline__ void async_ld_b128(unsigned lds_off, const void* g) {
  asm volatile("global_load_async_to_lds_b128 %0, %1, off"
               :: "v"(lds_off), "v"((unsigned long long)(uintptr_t)g)
               : "memory");
}
__device__ __forceinline__ void wait_async0() {
  asm volatile("s_wait_asynccnt 0x0" ::: "memory");
}
__device__ __forceinline__ void wait_async4() {
  asm volatile("s_wait_asynccnt 0x4" ::: "memory");
}

// Build a 16x32 f16 WMMA fragment (A layout; B layout is the same mapping with
// row = n). Lane l holds row (l&15); element i -> k = 16*(i>>3) + 8*(l>>4) + (i&7).
__device__ __forceinline__ v16h load_frag(const f16* __restrict__ base, int row,
                                          int ldk, int k0, int hi) {
  union { v16h v; h8 h[2]; } u;
  const f16* p = base + (size_t)row * ldk + k0 + 8 * hi;
  u.h[0] = *(const h8*)(p);
  u.h[1] = *(const h8*)(p + 16);
  return u.v;
}

// ---------------------------------------------------------------- conversions
__global__ void cvt_f32_f16(const float* __restrict__ in, f16* __restrict__ out, int n) {
  int i = blockIdx.x * blockDim.x + threadIdx.x;
  int stride = gridDim.x * blockDim.x;
  for (; i < n; i += stride) out[i] = (f16)in[i];
}

// V [T][HID] f32  ->  Vt [HID][T] f16   (row of Vt = h*HD + d)
__global__ void transpose_cvt(const float* __restrict__ V, f16* __restrict__ Vt,
                              int rows, int cols) {
  int i = blockIdx.x * blockDim.x + threadIdx.x;
  int n = rows * cols;
  int stride = gridDim.x * blockDim.x;
  for (; i < n; i += stride) {
    int t = i / cols, c = i - t * cols;
    Vt[(size_t)c * rows + t] = (f16)V[i];
  }
}

// ------------------------------------------------------------------ GEMM NT
// C[M][N] = A[M][K] @ W[N][K]^T.  8-wave block computes a 128x128 C tile.
// Per 32-k panel: async-copy 8KB A + 8KB B tiles to LDS (double-buffered),
// each wave does 8 WMMAs (4 m-tiles x 2 n-tiles) from 6 LDS fragments.
__global__ __launch_bounds__(256) void gemm_f16_nt_lds(
    const f16* __restrict__ A, const f16* __restrict__ W,
    float* __restrict__ C, int M, int N, int K) {
  __shared__ __align__(16) f16 smem[4 * 128 * LDP];
  f16* const sA0 = smem;
  f16* const sA1 = smem + 128 * LDP;
  f16* const sB0 = smem + 2 * 128 * LDP;
  f16* const sB1 = smem + 3 * 128 * LDP;

  const int tid  = threadIdx.x;
  const int lane = tid & 31;
  const int wave = tid >> 5;      // 0..7
  const int wm   = wave >> 2;     // 0..1  (64-row band)
  const int wn   = wave & 3;      // 0..3  (32-col band)
  const int lo = lane & 15, hi = lane >> 4;
  const int m0 = blockIdx.x * 128;
  const int n0 = blockIdx.y * 128;

  const int crow = tid >> 2;      // 0..63  copy row
  const int cseg = tid & 3;       // which 16B of the 64B k-row

  v8f acc[4][2] = {};
  const int KP = K / 32;

  auto issue = [&](int kp, f16* dA, f16* dB) {
    const f16* gA = A + (size_t)(m0 + crow) * K + kp * 32 + cseg * 8;
    const f16* gB = W + (size_t)(n0 + crow) * K + kp * 32 + cseg * 8;
    unsigned lA = (unsigned)(uintptr_t)(dA + crow * LDP + cseg * 8);
    unsigned lB = (unsigned)(uintptr_t)(dB + crow * LDP + cseg * 8);
    async_ld_b128(lA, gA);
    async_ld_b128(lA + 64 * LDP * 2, gA + (size_t)64 * K);
    async_ld_b128(lB, gB);
    async_ld_b128(lB + 64 * LDP * 2, gB + (size_t)64 * K);
  };

  issue(0, sA0, sB0);
  for (int kp = 0; kp < KP; ++kp) {
    f16* cA = (kp & 1) ? sA1 : sA0;
    f16* cB = (kp & 1) ? sB1 : sB0;
    if (kp + 1 < KP) {
      issue(kp + 1, (kp & 1) ? sA0 : sA1, (kp & 1) ? sB0 : sB1);
      wait_async4();   // current panel's 4 ops done; next panel in flight
    } else {
      wait_async0();
    }
    __syncthreads();
    v16h af[4], bf[2];
#pragma unroll
    for (int i = 0; i < 4; ++i) af[i] = load_frag(cA, wm * 64 + i * 16 + lo, LDP, 0, hi);
#pragma unroll
    for (int j = 0; j < 2; ++j) bf[j] = load_frag(cB, wn * 32 + j * 16 + lo, LDP, 0, hi);
#pragma unroll
    for (int i = 0; i < 4; ++i)
#pragma unroll
      for (int j = 0; j < 2; ++j)
        acc[i][j] = __builtin_amdgcn_wmma_f32_16x16x32_f16(
            false, af[i], false, bf[j], (short)0, acc[i][j], false, false);
    __syncthreads();   // all reads of this buffer done before it is re-filled
  }

#pragma unroll
  for (int i = 0; i < 4; ++i)
#pragma unroll
    for (int j = 0; j < 2; ++j)
#pragma unroll
      for (int r = 0; r < 8; ++r)
        C[(size_t)(m0 + wm * 64 + i * 16 + r + 8 * hi) * N +
          (n0 + wn * 32 + j * 16 + lo)] = acc[i][j][r];
}

// -------------------------------------------------------- ReBased feature map
// X [T][HID] f32 -> Y [NH][T][HD] f16 ; y = LN(x*gamma + beta) * outscale
__global__ __launch_bounds__(32) void feature_map_ln(
    const float* __restrict__ X, const float* __restrict__ gamma,
    const float* __restrict__ beta, f16* __restrict__ Y, float outscale) {
  const int t = blockIdx.x, h = blockIdx.y;
  const int lane = threadIdx.x;
  f4 xv = *(const f4*)(X + (size_t)t * HID + h * HD + lane * 4);
  f4 gv = *(const f4*)(gamma + lane * 4);
  f4 bv = *(const f4*)(beta + lane * 4);
  float y0 = xv.x * gv.x + bv.x;
  float y1 = xv.y * gv.y + bv.y;
  float y2 = xv.z * gv.z + bv.z;
  float y3 = xv.w * gv.w + bv.w;
  float s = y0 + y1 + y2 + y3;
#pragma unroll
  for (int m = 1; m < 32; m <<= 1) s += __shfl_xor(s, m, 32);
  float mu = s * (1.0f / 128.0f);
  float d0 = y0 - mu, d1 = y1 - mu, d2 = y2 - mu, d3 = y3 - mu;
  float vs = d0 * d0 + d1 * d1 + d2 * d2 + d3 * d3;
#pragma unroll
  for (int m = 1; m < 32; m <<= 1) vs += __shfl_xor(vs, m, 32);
  float inv = rsqrtf(vs * (1.0f / 128.0f) + LN_EPS) * outscale;
  h4 o;
  o.x = (f16)(d0 * inv); o.y = (f16)(d1 * inv);
  o.z = (f16)(d2 * inv); o.w = (f16)(d3 * inv);
  *(h4*)(Y + ((size_t)h * T_SEQ + t) * HD + lane * 4) = o;
}

// ------------------------------------------------------------ quadratic attn
__global__ __launch_bounds__(32) void rebased_attn(
    const f16* __restrict__ Qh,  // [NH][T][HD]
    const f16* __restrict__ Kh,  // [NH][T][HD]
    const f16* __restrict__ Vt,  // [HID][T]  (row = h*HD + d)
    f16* __restrict__ Oh,        // [T][HID]
    int T) {
  __shared__ __align__(16) f16 sS[16 * 32];
  const int lane = threadIdx.x;
  const int lo = lane & 15, hi = lane >> 4;
  const int q0 = blockIdx.x * 16;
  const int h = blockIdx.y;
  const f16* Qb = Qh + (size_t)h * T * HD;
  const f16* Kb = Kh + (size_t)h * T * HD;
  const f16* Vb = Vt + (size_t)h * HD * T;

  v16h aq[4];
#pragma unroll
  for (int kc = 0; kc < 4; ++kc) aq[kc] = load_frag(Qb, q0 + lo, HD, 32 * kc, hi);

  v8f oacc[8] = {};
  float zp[8] = {};

  for (int kv0 = 0; kv0 < q0 + 16; kv0 += 32) {
    __syncthreads();
#pragma unroll
    for (int jt = 0; jt < 2; ++jt) {
      v8f s = {};
#pragma unroll
      for (int kc = 0; kc < 4; ++kc) {
        v16h bk = load_frag(Kb, kv0 + 16 * jt + lo, HD, 32 * kc, hi);
        s = __builtin_amdgcn_wmma_f32_16x16x32_f16(
            false, aq[kc], false, bk, (short)0, s, false, false);
      }
      const int key = kv0 + 16 * jt + lo;
#pragma unroll
      for (int r = 0; r < 8; ++r) {
        const int qr = q0 + r + 8 * hi;
        float sv = s[r];
        float s2 = (key <= qr) ? sv * sv : 0.0f;
        zp[r] += s2;
        sS[(r + 8 * hi) * 32 + 16 * jt + lo] = (f16)s2;
      }
    }
    __syncthreads();
    v16h as = load_frag(sS, lo, 32, 0, hi);
#pragma unroll
    for (int j = 0; j < 8; ++j) {
      v16h bv = load_frag(Vb, 16 * j + lo, T, kv0, hi);
      oacc[j] = __builtin_amdgcn_wmma_f32_16x16x32_f16(
          false, as, false, bv, (short)0, oacc[j], false, false);
    }
  }

#pragma unroll
  for (int r = 0; r < 8; ++r) {
#pragma unroll
    for (int m = 1; m < 16; m <<= 1) zp[r] += __shfl_xor(zp[r], m, 32);
  }
#pragma unroll
  for (int j = 0; j < 8; ++j)
#pragma unroll
    for (int r = 0; r < 8; ++r) {
      float val = oacc[j][r] / (zp[r] + LN_EPS);
      Oh[(size_t)(q0 + r + 8 * hi) * HID + h * HD + 16 * j + lo] = (f16)val;
    }
}

// ----------------------------------------------------------------- launcher
extern "C" void kernel_launch(void* const* d_in, const int* in_sizes, int n_in,
                              void* d_out, int out_size, void* d_ws, size_t ws_size,
                              hipStream_t stream) {
  (void)in_sizes; (void)n_in; (void)out_size; (void)ws_size;
  const float* X     = (const float*)d_in[0];
  const float* Wq    = (const float*)d_in[1];
  const float* Wk    = (const float*)d_in[2];
  const float* Wv    = (const float*)d_in[3];
  const float* Wo    = (const float*)d_in[4];
  const float* gamma = (const float*)d_in[5];
  const float* beta  = (const float*)d_in[6];
  float* out = (float*)d_out;

  char* ws = (char*)d_ws;
  size_t off = 0;
  auto alloc = [&](size_t bytes) -> char* {
    char* p = ws + off;
    off += (bytes + 255) & ~(size_t)255;
    return p;
  };
  const size_t NEL = (size_t)T_SEQ * HID;
  f16*   Xh  = (f16*)alloc(NEL * 2);
  f16*   Wqh = (f16*)alloc(NEL * 2);
  f16*   Wkh = (f16*)alloc(NEL * 2);
  f16*   Wvh = (f16*)alloc(NEL * 2);
  f16*   Woh = (f16*)alloc(NEL * 2);
  float* Qf  = (float*)alloc(NEL * 4);
  float* Kf  = (float*)alloc(NEL * 4);
  float* Vf  = (float*)alloc(NEL * 4);
  f16*   Qhm = (f16*)alloc(NEL * 2);
  f16*   Khm = (f16*)alloc(NEL * 2);
  f16*   Vtm = (f16*)alloc(NEL * 2);
  f16*   Ohm = (f16*)alloc(NEL * 2);

  // 1) f32 -> f16 copies
  cvt_f32_f16<<<2048, 256, 0, stream>>>(X,  Xh,  (int)NEL);
  cvt_f32_f16<<<2048, 256, 0, stream>>>(Wq, Wqh, (int)NEL);
  cvt_f32_f16<<<2048, 256, 0, stream>>>(Wk, Wkh, (int)NEL);
  cvt_f32_f16<<<2048, 256, 0, stream>>>(Wv, Wvh, (int)NEL);
  cvt_f32_f16<<<2048, 256, 0, stream>>>(Wo, Woh, (int)NEL);

  // 2) Q/K/V projections (x @ W^T)
  dim3 ggrid(T_SEQ / 128, HID / 128);
  gemm_f16_nt_lds<<<ggrid, 256, 0, stream>>>(Xh, Wqh, Qf, T_SEQ, HID, HID);
  gemm_f16_nt_lds<<<ggrid, 256, 0, stream>>>(Xh, Wkh, Kf, T_SEQ, HID, HID);
  gemm_f16_nt_lds<<<ggrid, 256, 0, stream>>>(Xh, Wvh, Vf, T_SEQ, HID, HID);

  // 3) feature maps (q gets the D^-0.5 scale folded in), V transpose
  feature_map_ln<<<dim3(T_SEQ, NH), 32, 0, stream>>>(Qf, gamma, beta, Qhm,
                                                     0.08838834764831845f);
  feature_map_ln<<<dim3(T_SEQ, NH), 32, 0, stream>>>(Kf, gamma, beta, Khm, 1.0f);
  transpose_cvt<<<2048, 256, 0, stream>>>(Vf, Vtm, T_SEQ, HID);

  // 4) causal quadratic attention
  rebased_attn<<<dim3(T_SEQ / 16, NH), 32, 0, stream>>>(Qhm, Khm, Vtm, Ohm, T_SEQ);

  // 5) output projection
  gemm_f16_nt_lds<<<ggrid, 256, 0, stream>>>(Ohm, Woh, out, T_SEQ, HID, HID);
}
